// RemixedBlock_3341484556518
// MI455X (gfx1250) — compile-verified, matrix operand order checked by reading, not run on hardware
//
#include <hip/hip_runtime.h>

typedef unsigned short u16;
typedef unsigned int   u32;
typedef __attribute__((ext_vector_type(16))) __bf16 v16bf;
typedef __attribute__((ext_vector_type(8)))  __bf16 v8bf;
typedef __attribute__((ext_vector_type(8)))  float  v8f;
typedef __attribute__((ext_vector_type(4)))  u32    u32x4;
typedef __attribute__((ext_vector_type(4)))  int    i32x4;
typedef __attribute__((ext_vector_type(8)))  int    i32x8;

#if __has_builtin(__builtin_amdgcn_tensor_load_to_lds)
#define USE_TDM 1
#else
#define USE_TDM 0
#endif

// ---------------- constants (match reference) ----------------
#define Bc   2
#define Tc   2048
#define Ec   1024
#define Hc   16
#define Dc   64
#define Cc   64
#define Pc   64
#define BFc  256
#define MHc  128
#define Rc   8
#define F4c  4096
#define NTOK (Bc*Tc)   // 4096

// epilogue codes
#define EP_F32        0  // (+optional bias) -> f32
#define EP_BF16       1  // -> bf16
#define EP_BIAS_GELU  2  // +bias, gelu -> bf16
#define EP_RESID      4  // +resid -> f32
#define EP_GATE_FC    5  // *gate + bias, relu^2 -> bf16
#define EP_GATE_PJ    6  // *gate + bias, +resid -> f32

__device__ __forceinline__ u16 f2bfu(float f) {
  unsigned u = __float_as_uint(f);
  unsigned r = (u + 0x7fffu + ((u >> 16) & 1u)) >> 16;
  return (u16)r;
}
__device__ __forceinline__ float geluf(float x) {
  return 0.5f * x * (1.0f + erff(x * 0.70710678118654752f));
}

// ---- Tensor Data Mover: 2-D bf16 tile (tile_d1 rows x tile_d0 cols) -> LDS ----
// D# layout per CDNA5 ISA ch.8: group0 {count,lds_addr,global_addr,type=2},
// group1 {data_size=2B, tensor_dim0/1, tile_dim0/1, tensor_dim0_stride}.
__device__ __forceinline__ void tdm_load_2d(u32 ldsaddr, const void* gaddr,
                                            u32 tensor_d0, u32 tensor_d1, u32 stride0,
                                            u32 tile_d0, u32 tile_d1) {
#if USE_TDM
  unsigned long long ga = (unsigned long long)(size_t)gaddr;
  u32x4 g0;
  g0[0] = 1u;                                        // count=1, user descriptor
  g0[1] = ldsaddr;                                   // lds_addr [63:32]
  g0[2] = (u32)(ga & 0xffffffffu);                   // global_addr[31:0] -> bits 95:64
  g0[3] = (u32)((ga >> 32) & 0x01ffffffu) | (2u << 30); // addr[56:32] + type=2
  i32x8 g1;
  g1[0] = (int)(1u << 16);                           // workgroup_mask=0, data_size=1 (2B)
  g1[1] = (int)((tensor_d0 & 0xffffu) << 16);        // tensor_dim0[15:0] at bits 63:48
  g1[2] = (int)((tensor_d0 >> 16) | ((tensor_d1 & 0xffffu) << 16)); // td0 hi, td1 lo
  g1[3] = (int)((tensor_d1 >> 16) | (tile_d0 << 16));               // td1 hi, tile_dim0
  g1[4] = (int)(tile_d1 & 0xffffu);                  // tile_dim1, tile_dim2=0
  g1[5] = (int)stride0;                              // tensor_dim0_stride[31:0]
  g1[6] = 0;                                         // stride hi, dim1_stride lo
  g1[7] = 0;
  i32x4 z4 = {0, 0, 0, 0};
#if defined(__clang_major__) && (__clang_major__ >= 23)
  i32x8 z8 = {0, 0, 0, 0, 0, 0, 0, 0};
  __builtin_amdgcn_tensor_load_to_lds(g0, g1, z4, z4, z8, 0);
#else
  __builtin_amdgcn_tensor_load_to_lds(g0, g1, z4, z4, 0);
#endif
#else
  (void)ldsaddr; (void)gaddr; (void)tensor_d0; (void)tensor_d1;
  (void)stride0; (void)tile_d0; (void)tile_d1;
#endif
}

// Build a 16x32 bf16 A/B fragment per CDNA5 ISA layout:
// lane&15 = row, lane>>4 = half; chunk0 at kbase+8*half, chunk1 at +16 more.
__device__ __forceinline__ v16bf frag_ld(const u16* p, int row, int stride, int kbase, int half) {
  const __bf16* q = (const __bf16*)(p + (size_t)row * stride + kbase + 8 * half);
  v8bf lo = *(const v8bf*)q;
  v8bf hi = *(const v8bf*)(q + 16);
  v16bf r;
#pragma unroll
  for (int e = 0; e < 8; ++e) { r[e] = lo[e]; r[e + 8] = hi[e]; }
  return r;
}
__device__ __forceinline__ v8f wmma_bf16(v16bf a, v16bf b, v8f c) {
  return __builtin_amdgcn_wmma_f32_16x16x32_bf16(false, a, false, b, (short)0, c, false, false);
}

// ---------------- f32 -> bf16 convert ----------------
__global__ __launch_bounds__(256) void k_cvt(const float* __restrict__ s, u16* __restrict__ d, int n) {
  int i = blockIdx.x * 256 + threadIdx.x;
  if (i < n) d[i] = f2bfu(s[i]);
}

// ---------------- RMS norm -> bf16 ----------------
__global__ __launch_bounds__(256) void k_rms(const float* __restrict__ x, u16* __restrict__ ob,
                                             int cols, float eps) {
  __shared__ float red[8];
  int row = blockIdx.x, t = threadIdx.x;
  const float* xr = x + (size_t)row * cols;
  float ss = 0.f;
  for (int c = t; c < cols; c += 256) { float v = xr[c]; ss += v * v; }
#pragma unroll
  for (int m = 1; m < 32; m <<= 1) ss += __shfl_xor(ss, m, 32);
  if ((t & 31) == 0) red[t >> 5] = ss;
  __syncthreads();
  float tot = 0.f;
#pragma unroll
  for (int i = 0; i < 8; ++i) tot += red[i];
  float sc = rsqrtf(tot / (float)cols + eps);
  u16* po = ob + (size_t)row * cols;
  for (int c = t; c < cols; c += 256) po[c] = f2bfu(xr[c] * sc);
}

// ---------------- rotary + pack to (B,H,T,D) bf16 ----------------
__global__ __launch_bounds__(64) void k_rotary(const float* __restrict__ qf, const float* __restrict__ kf,
                                               const float* __restrict__ vf,
                                               const float* __restrict__ cosp, const float* __restrict__ sinp,
                                               u16* __restrict__ qb, u16* __restrict__ kb, u16* __restrict__ vb) {
  int idx = blockIdx.x;            // over B*T*H
  int d = threadIdx.x;             // 0..63
  int h = idx % Hc;
  int bt = idx / Hc;
  int tpos = bt % Tc;
  int b = bt / Tc;
  size_t src = (size_t)bt * Ec + h * Dc;
  size_t dst = (((size_t)(b * Hc + h)) * Tc + tpos) * Dc;
  int d2 = d & 31;
  float c_ = cosp[tpos * 32 + d2], s_ = sinp[tpos * 32 + d2];
  float q1 = qf[src + d2], q2 = qf[src + d2 + 32];
  float k1 = kf[src + d2], k2 = kf[src + d2 + 32];
  float qv, kv;
  if (d < 32) { qv = q1 * c_ + q2 * s_;  kv = k1 * c_ + k2 * s_; }
  else        { qv = -q1 * s_ + q2 * c_; kv = -k1 * s_ + k2 * c_; }
  qb[dst + d] = f2bfu(qv);
  kb[dst + d] = f2bfu(kv);
  vb[dst + d] = f2bfu(vf[src + d]);
}

// ---------------- flash attention (causal), bf16 WMMA ----------------
// grid: (B*H, T/128); block 256 (8 waves). Each wave owns 16 q rows.
__global__ __launch_bounds__(256) void k_attn(const u16* __restrict__ qb, const u16* __restrict__ kb,
                                              const u16* __restrict__ vb, u16* __restrict__ out) {
  __shared__ u16 lK[64 * 64];        // [key][d]
  __shared__ u16 lV[64 * 64];        // transposed: [d][key]
  __shared__ u16 lP[128 * 64];       // probs strip per wave
  int bh = blockIdx.x;
  int b = bh / Hc, h = bh % Hc;
  int qbase = blockIdx.y * 128;
  int t = threadIdx.x, wave = t >> 5, lane = t & 31, half = lane >> 4, ln = lane & 15;
  const u16* Q = qb + (size_t)bh * Tc * Dc;
  const u16* Kp = kb + (size_t)bh * Tc * Dc;
  const u16* Vp = vb + (size_t)bh * Tc * Dc;
  int qrow0 = qbase + wave * 16;
  const u16* Qrow = Q + (size_t)(qrow0 + ln) * Dc;
  v16bf aq0 = frag_ld(Qrow, 0, 0, 0, half);
  v16bf aq1 = frag_ld(Qrow, 0, 0, 32, half);
  v8f o[4];
  float m[8], l[8];
#pragma unroll
  for (int f = 0; f < 4; ++f) o[f] = (v8f){0.f,0.f,0.f,0.f,0.f,0.f,0.f,0.f};
#pragma unroll
  for (int vi = 0; vi < 8; ++vi) { m[vi] = -3e38f; l[vi] = 0.f; }

  int nkt = blockIdx.y * 2 + 2;      // tiles with kb0 <= qbase+64
  for (int kt = 0; kt < nkt; ++kt) {
    int kb0 = kt * 64;
    __syncthreads();
#if USE_TDM
    if (t == 0) {  // K tile via Tensor Data Mover (64x64 bf16, row-major)
      tdm_load_2d((u32)(size_t)(void*)lK, Kp + (size_t)kb0 * Dc,
                  Dc, Tc, Dc, 64, 64);
    }
#endif
    { // V tile: manual transposed store (TDM has no transpose)
      int r = t >> 2, c = (t & 3) * 16;
#if !USE_TDM
      *(uint4*)&lK[r * 64 + c]     = *(const uint4*)&Kp[(size_t)(kb0 + r) * Dc + c];
      *(uint4*)&lK[r * 64 + c + 8] = *(const uint4*)&Kp[(size_t)(kb0 + r) * Dc + c + 8];
#endif
      uint4 v0 = *(const uint4*)&Vp[(size_t)(kb0 + r) * Dc + c];
      uint4 v1 = *(const uint4*)&Vp[(size_t)(kb0 + r) * Dc + c + 8];
      const u16* pv0 = (const u16*)&v0;
      const u16* pv1 = (const u16*)&v1;
#pragma unroll
      for (int e = 0; e < 8; ++e) {
        lV[(c + e) * 64 + r]     = pv0[e];
        lV[(c + 8 + e) * 64 + r] = pv1[e];
      }
    }
#if USE_TDM
    if (t == 0) __builtin_amdgcn_s_wait_tensorcnt(0);
#endif
    __syncthreads();
    if (kb0 <= qrow0 + 15) {        // causal: this wave needs this key tile
      v8f s[4];
#pragma unroll
      for (int f = 0; f < 4; ++f) s[f] = (v8f){0.f,0.f,0.f,0.f,0.f,0.f,0.f,0.f};
#pragma unroll
      for (int ds_ = 0; ds_ < 2; ++ds_) {
        v16bf aq = ds_ ? aq1 : aq0;
#pragma unroll
        for (int f = 0; f < 4; ++f) {
          v16bf bk = frag_ld(lK, f * 16 + ln, 64, ds_ * 32, half);
          s[f] = wmma_bf16(aq, bk, s[f]);
        }
      }
      // online softmax on the 16x64 strip
#pragma unroll
      for (int vi = 0; vi < 8; ++vi) {
        int r = qrow0 + vi + 8 * half;
        float mx = -3e38f;
#pragma unroll
        for (int f = 0; f < 4; ++f) {
          int c = kb0 + f * 16 + ln;
          float v = s[f][vi] * 0.125f;          // D^-0.5
          if (c > r) v = -3e38f;
          s[f][vi] = v;
          mx = fmaxf(mx, v);
        }
#pragma unroll
        for (int mk = 1; mk < 16; mk <<= 1) mx = fmaxf(mx, __shfl_xor(mx, mk, 32));
        float mn = fmaxf(m[vi], mx);
        float alpha = __expf(m[vi] - mn);
        m[vi] = mn;
        float rs = 0.f;
#pragma unroll
        for (int f = 0; f < 4; ++f) {
          float p = __expf(s[f][vi] - mn);
          s[f][vi] = p;
          rs += p;
        }
#pragma unroll
        for (int mk = 1; mk < 16; mk <<= 1) rs += __shfl_xor(rs, mk, 32);
        l[vi] = l[vi] * alpha + rs;
#pragma unroll
        for (int f = 0; f < 4; ++f) o[f][vi] *= alpha;
        int pr = wave * 16 + vi + 8 * half;
#pragma unroll
        for (int f = 0; f < 4; ++f) lP[pr * 64 + f * 16 + ln] = f2bfu(s[f][vi]);
      }
      // O += P x V
#pragma unroll
      for (int ks = 0; ks < 2; ++ks) {
        v16bf ap = frag_ld(lP + wave * 16 * 64, ln, 64, ks * 32, half);
#pragma unroll
        for (int f = 0; f < 4; ++f) {
          v16bf bv = frag_ld(lV, f * 16 + ln, 64, ks * 32, half);
          o[f] = wmma_bf16(ap, bv, o[f]);
        }
      }
    }
  }
  // normalize + store bf16 into (B,T,E) at column h*64
#pragma unroll
  for (int vi = 0; vi < 8; ++vi) {
    int r = qrow0 + vi + 8 * half;
    float inv = 1.f / l[vi];
    size_t base = ((size_t)(b * Tc + r)) * Ec + h * Dc;
#pragma unroll
    for (int f = 0; f < 4; ++f) out[base + f * 16 + ln] = f2bfu(o[f][vi] * inv);
  }
}

// ---------------- tiled bf16 WMMA GEMM: Out[i][j] = sum_k A[i][k]*W[j][k] ----------------
// grid: (M/64, N/64); block 256 (8 waves), K multiple of 32. Tiles staged by TDM.
__global__ __launch_bounds__(256) void k_gemm(const u16* __restrict__ A, const u16* __restrict__ W,
                                              int K, int M,
                                              float* __restrict__ outF, u16* __restrict__ outB,
                                              const float* __restrict__ bias,
                                              const float* __restrict__ resid,
                                              const float* __restrict__ og8,
                                              const float* __restrict__ ogB,
                                              const float* __restrict__ ogs, int ep) {
  __shared__ u16 lA[64 * 32];
  __shared__ u16 lB[64 * 32];
  int i0 = blockIdx.y * 64, j0 = blockIdx.x * 64;
  int Ntot = gridDim.y * 64;
  int t = threadIdx.x, wave = t >> 5, lane = t & 31, half = lane >> 4, ln = lane & 15;
  int wi = wave >> 1, wj = wave & 1;
  v8f c0 = (v8f){0.f,0.f,0.f,0.f,0.f,0.f,0.f,0.f};
  v8f c1 = c0;
  int ldrow = t >> 2, ldcol = (t & 3) * 8;
  const u16* pA = A + (size_t)(i0 + ldrow) * K + ldcol;
  const u16* pW = W + (size_t)(j0 + ldrow) * K + ldcol;
  for (int k0 = 0; k0 < K; k0 += 32) {
    __syncthreads();
#if USE_TDM
    if (t == 0) {  // stage both 64x32 bf16 tiles via Tensor Data Mover
      tdm_load_2d((u32)(size_t)(void*)lA, A + (size_t)i0 * K + k0, K, (u32)Ntot, K, 32, 64);
      tdm_load_2d((u32)(size_t)(void*)lB, W + (size_t)j0 * K + k0, K, (u32)M,    K, 32, 64);
      __builtin_amdgcn_s_wait_tensorcnt(0);
    }
#else
    *(uint4*)&lA[ldrow * 32 + ldcol] = *(const uint4*)(pA + k0);
    *(uint4*)&lB[ldrow * 32 + ldcol] = *(const uint4*)(pW + k0);
#endif
    if (k0 + 32 < K) {
      __builtin_prefetch(pA + k0 + 32, 0, 1);   // global_prefetch_b8
      __builtin_prefetch(pW + k0 + 32, 0, 1);
    }
    __syncthreads();
    v16bf a  = frag_ld(lA, wi * 16 + ln, 32, 0, half);
    v16bf b0 = frag_ld(lB, wj * 32 + ln, 32, 0, half);
    v16bf b1 = frag_ld(lB, wj * 32 + 16 + ln, 32, 0, half);
    c0 = wmma_bf16(a, b0, c0);
    c1 = wmma_bf16(a, b1, c1);
  }
  int rbase = i0 + wi * 16 + 8 * half;
  int cb0 = j0 + wj * 32 + ln;
  int cb1 = cb0 + 16;
  float scal = (ep == EP_GATE_FC || ep == EP_GATE_PJ) ? ogs[0] : 0.f;

#pragma unroll
  for (int vi = 0; vi < 8; ++vi) {
    int r = rbase + vi;
#pragma unroll
    for (int e = 0; e < 2; ++e) {
      int c = e ? cb1 : cb0;
      float acc = e ? c1[vi] : c0[vi];
      size_t oi = (size_t)r * M + c;
      if (ep == EP_F32) {
        if (bias) acc += bias[c];
        outF[oi] = acc;
      } else if (ep == EP_BF16) {
        outB[oi] = f2bfu(acc);
      } else if (ep == EP_BIAS_GELU) {
        outB[oi] = f2bfu(geluf(acc + bias[c]));
      } else if (ep == EP_RESID) {
        outF[oi] = acc + resid[oi];
      } else { // gate epilogues
        float s = 0.f;
        const float* o8 = og8 + (size_t)r * 8;
#pragma unroll
        for (int q = 0; q < 8; ++q) s += o8[q] * ogB[(size_t)q * M + c];
        float gate = 1.f + tanhf(scal * s);
        float v = acc * gate + bias[c];
        if (ep == EP_GATE_FC) {
          float rl = fmaxf(v, 0.f);
          outB[oi] = f2bfu(rl * rl);
        } else {
          outF[oi] = resid[oi] + v;
        }
      }
    }
  }
}

// ---------------- proto softmax context: cq -> w -> ctx ----------------
__global__ __launch_bounds__(64) void k_ctx(const float* __restrict__ cq, const float* __restrict__ protos,
                                            float* __restrict__ ctxf, u16* __restrict__ ctxb) {
  __shared__ float lc[64];
  __shared__ float lw[64];
  int row = blockIdx.x, t = threadIdx.x;
  lc[t] = cq[(size_t)row * Cc + t];
  __syncthreads();
  float s = 0.f;
  for (int c = 0; c < 64; ++c) s += lc[c] * protos[t * 64 + c];
  s *= 0.125f;                        // C^-0.5
  lw[t] = s; __syncthreads();
  float mx = -3e38f;
  for (int p = 0; p < 64; ++p) mx = fmaxf(mx, lw[p]);
  float e = __expf(s - mx);
  __syncthreads(); lw[t] = e; __syncthreads();
  float sum = 0.f;
  for (int p = 0; p < 64; ++p) sum += lw[p];
  float w = e / sum;
  __syncthreads(); lw[t] = w; __syncthreads();
  float cv = 0.f;
  for (int p = 0; p < 64; ++p) cv += lw[p] * protos[p * 64 + t];
  ctxf[(size_t)row * Cc + t] = cv;
  ctxb[(size_t)row * Cc + t] = f2bfu(cv);
}

// ---------------- og8 = ctx @ ogw.T + ogb ----------------
__global__ __launch_bounds__(256) void k_og(const float* __restrict__ ctx, const float* __restrict__ ogw,
                                            const float* __restrict__ ogb, float* __restrict__ og8) {
  int i = blockIdx.x * 256 + threadIdx.x;
  if (i >= NTOK * Rc) return;
  int row = i >> 3, r = i & 7;
  const float* cr = ctx + (size_t)row * Cc;
  const float* wr = ogw + r * Cc;
  float s = ogb[r];
  for (int c = 0; c < 64; ++c) s += cr[c] * wr[c];
  og8[i] = s;
}

// ---------------- LN(t1) * sigmoid(gl) -> bf16 ----------------
__global__ __launch_bounds__(256) void k_ln_gate(const float* __restrict__ t1, const float* __restrict__ gl,
                                                 const float* __restrict__ g, const float* __restrict__ bb,
                                                 u16* __restrict__ ob) {
  __shared__ float red[8];
  int row = blockIdx.x, t = threadIdx.x;
  size_t idx = (size_t)row * BFc + t;
  float v = t1[idx];
  float s = v;
#pragma unroll
  for (int mk = 1; mk < 32; mk <<= 1) s += __shfl_xor(s, mk, 32);
  if ((t & 31) == 0) red[t >> 5] = s;
  __syncthreads();
  float mu = 0.f;
#pragma unroll
  for (int i = 0; i < 8; ++i) mu += red[i];
  mu *= (1.f / 256.f);
  __syncthreads();
  float dv = v - mu, s2 = dv * dv;
#pragma unroll
  for (int mk = 1; mk < 32; mk <<= 1) s2 += __shfl_xor(s2, mk, 32);
  if ((t & 31) == 0) red[t >> 5] = s2;
  __syncthreads();
  float var = 0.f;
#pragma unroll
  for (int i = 0; i < 8; ++i) var += red[i];
  var *= (1.f / 256.f);
  float y = dv * rsqrtf(var + 1e-5f) * g[t] + bb[t];
  float gb = 1.f / (1.f + __expf(-gl[idx]));
  ob[idx] = f2bfu(y * gb);
}

// ================= host side =================
extern "C" void kernel_launch(void* const* d_in, const int* in_sizes, int n_in,
                              void* d_out, int out_size, void* d_ws, size_t ws_size,
                              hipStream_t stream) {
  (void)in_sizes; (void)n_in; (void)out_size; (void)ws_size;
  const float* x     = (const float*)d_in[0];
  const float* cosp  = (const float*)d_in[1];
  const float* sinp  = (const float*)d_in[2];
  const float* wq    = (const float*)d_in[3];
  const float* wk    = (const float*)d_in[4];
  const float* wv    = (const float*)d_in[5];
  const float* wo    = (const float*)d_in[6];
  const float* cbqw  = (const float*)d_in[7];
  const float* protos= (const float*)d_in[8];
  const float* fc_bw = (const float*)d_in[9];
  const float* fc_lg = (const float*)d_in[10];
  const float* fc_lb = (const float*)d_in[11];
  const float* fc_tm = (const float*)d_in[12];
  const float* fc_b  = (const float*)d_in[13];
  const float* fc_m1w= (const float*)d_in[14];
  const float* fc_m1b= (const float*)d_in[15];
  const float* fc_m2w= (const float*)d_in[16];
  const float* fc_m2b= (const float*)d_in[17];
  const float* fc_ogw= (const float*)d_in[18];
  const float* fc_ogb= (const float*)d_in[19];
  const float* fc_ogB= (const float*)d_in[20];
  const float* fc_ogs= (const float*)d_in[21];
  const float* pj_bw = (const float*)d_in[22];
  const float* pj_lg = (const float*)d_in[23];
  const float* pj_lb = (const float*)d_in[24];
  const float* pj_tm = (const float*)d_in[25];
  const float* pj_b  = (const float*)d_in[26];
  const float* pj_m1w= (const float*)d_in[27];
  const float* pj_m1b= (const float*)d_in[28];
  const float* pj_m2w= (const float*)d_in[29];
  const float* pj_m2b= (const float*)d_in[30];
  const float* pj_ogw= (const float*)d_in[31];
  const float* pj_ogb= (const float*)d_in[32];
  const float* pj_ogB= (const float*)d_in[33];
  const float* pj_ogs= (const float*)d_in[34];

  char* ws = (char*)d_ws;
  const size_t MB = (size_t)1 << 20;
  // phase-1 arena
  u16*   xnb  = (u16*)  (ws + 0 * MB);     // 8MB
  float* qf   = (float*)(ws + 8 * MB);     // 16MB
  float* kf   = (float*)(ws + 24 * MB);    // 16MB
  float* vf   = (float*)(ws + 40 * MB);    // 16MB
  u16*   qb   = (u16*)  (ws + 56 * MB);    // 8MB
  u16*   kb   = (u16*)  (ws + 64 * MB);    // 8MB
  u16*   vb   = (u16*)  (ws + 72 * MB);    // 8MB
  u16*   attn = (u16*)  (ws + 80 * MB);    // 8MB
  float* x2   = (float*)(ws + 88 * MB);    // 16MB
  u16*   xn2b = (u16*)  (ws + 104 * MB);   // 8MB
  // phase-2 arena (reuses qf/kf/vf space, valid after attention)
  u16*   hB   = (u16*)  (ws + 8 * MB);     // 32MB  [8,40)
  float* t1   = (float*)(ws + 40 * MB);    // 4MB
  float* gl   = (float*)(ws + 44 * MB);    // 4MB
  u16*   m1oB = (u16*)  (ws + 48 * MB);    // 1MB
  float* og8  = (float*)(ws + 49 * MB);    // 128KB
  u16*   atmB = (u16*)  (ws + 50 * MB);    // 2MB
  float* cq   = (float*)(ws + 52 * MB);    // 1MB
  float* ctxf = (float*)(ws + 53 * MB);    // 1MB
  u16*   ctxb = (u16*)  (ws + 54 * MB);    // 0.5MB
  // bf16 weights
  char* wp = ws + 112 * MB;
  auto walloc = [&](int n) { u16* p = (u16*)wp; wp += (size_t)n * 2; return p; };
  u16* wqB  = walloc(Ec * Ec);
  u16* wkB  = walloc(Ec * Ec);
  u16* wvB  = walloc(Ec * Ec);
  u16* woB  = walloc(Ec * Ec);
  u16* cbB  = walloc(Cc * Ec);
  u16* fbwB = walloc(BFc * Ec);
  u16* ftmB = walloc(F4c * BFc);
  u16* fm1B = walloc(MHc * Cc);
  u16* fm2B = walloc(BFc * MHc);
  u16* pbwB = walloc(BFc * F4c);
  u16* ptmB = walloc(Ec * BFc);
  u16* pm1B = walloc(MHc * Cc);
  u16* pm2B = walloc(BFc * MHc);

  auto cvt = [&](const float* s, u16* d, int n) {
    k_cvt<<<dim3((n + 255) / 256), dim3(256), 0, stream>>>(s, d, n);
  };
  cvt(wq, wqB, Ec * Ec);   cvt(wk, wkB, Ec * Ec);
  cvt(wv, wvB, Ec * Ec);   cvt(wo, woB, Ec * Ec);
  cvt(cbqw, cbB, Cc * Ec); cvt(fc_bw, fbwB, BFc * Ec);
  cvt(fc_tm, ftmB, F4c * BFc);
  cvt(fc_m1w, fm1B, MHc * Cc); cvt(fc_m2w, fm2B, BFc * MHc);
  cvt(pj_bw, pbwB, BFc * F4c); cvt(pj_tm, ptmB, Ec * BFc);
  cvt(pj_m1w, pm1B, MHc * Cc); cvt(pj_m2w, pm2B, BFc * MHc);

  auto gemm = [&](const u16* A, const u16* W, int K, int M,
                  float* oF, u16* oB, const float* bias, const float* resid,
                  const float* g8, const float* gB, const float* gs, int ep) {
    k_gemm<<<dim3(M / 64, NTOK / 64), dim3(256), 0, stream>>>(A, W, K, M, oF, oB,
                                                              bias, resid, g8, gB, gs, ep);
  };

  // 1) rms(x) -> bf16
  k_rms<<<dim3(NTOK), dim3(256), 0, stream>>>(x, xnb, Ec, 1e-6f);
  // 2) q,k,v projections (f32 out)
  gemm(xnb, wqB, Ec, Ec, qf, nullptr, nullptr, nullptr, nullptr, nullptr, nullptr, EP_F32);
  gemm(xnb, wkB, Ec, Ec, kf, nullptr, nullptr, nullptr, nullptr, nullptr, nullptr, EP_F32);
  gemm(xnb, wvB, Ec, Ec, vf, nullptr, nullptr, nullptr, nullptr, nullptr, nullptr, EP_F32);
  // 3) rotary + relayout to (B,H,T,D) bf16
  k_rotary<<<dim3(Bc * Tc * Hc), dim3(64), 0, stream>>>(qf, kf, vf, cosp, sinp, qb, kb, vb);
  // 4) flash attention
  k_attn<<<dim3(Bc * Hc, Tc / 128), dim3(256), 0, stream>>>(qb, kb, vb, attn);
  // 5) x2 = x + attn @ wo.T
  gemm(attn, woB, Ec, Ec, x2, nullptr, nullptr, x, nullptr, nullptr, nullptr, EP_RESID);
  // 6) rms(x2) -> bf16
  k_rms<<<dim3(NTOK), dim3(256), 0, stream>>>(x2, xn2b, Ec, 1e-6f);
  // 7) cq = xn2 @ cb_qw.T ; ctx via proto softmax
  gemm(xn2b, cbB, Ec, Cc, cq, nullptr, nullptr, nullptr, nullptr, nullptr, nullptr, EP_F32);
  k_ctx<<<dim3(NTOK), dim3(64), 0, stream>>>(cq, protos, ctxf, ctxb);

  // ---- fc remixed ----
  gemm(xn2b, fbwB, Ec, BFc, t1, nullptr, nullptr, nullptr, nullptr, nullptr, nullptr, EP_F32);
  gemm(ctxb, fm1B, Cc, MHc, nullptr, m1oB, fc_m1b, nullptr, nullptr, nullptr, nullptr, EP_BIAS_GELU);
  gemm(m1oB, fm2B, MHc, BFc, gl, nullptr, fc_m2b, nullptr, nullptr, nullptr, nullptr, EP_F32);
  k_og<<<dim3(NTOK * Rc / 256), dim3(256), 0, stream>>>(ctxf, fc_ogw, fc_ogb, og8);
  k_ln_gate<<<dim3(NTOK), dim3(256), 0, stream>>>(t1, gl, fc_lg, fc_lb, atmB);
  gemm(atmB, ftmB, BFc, F4c, nullptr, hB, fc_b, nullptr, og8, fc_ogB, fc_ogs, EP_GATE_FC);

  // ---- pj remixed ----
  gemm(hB, pbwB, F4c, BFc, t1, nullptr, nullptr, nullptr, nullptr, nullptr, nullptr, EP_F32);
  gemm(ctxb, pm1B, Cc, MHc, nullptr, m1oB, pj_m1b, nullptr, nullptr, nullptr, nullptr, EP_BIAS_GELU);
  gemm(m1oB, pm2B, MHc, BFc, gl, nullptr, pj_m2b, nullptr, nullptr, nullptr, nullptr, EP_F32);
  k_og<<<dim3(NTOK * Rc / 256), dim3(256), 0, stream>>>(ctxf, pj_ogw, pj_ogb, og8);
  k_ln_gate<<<dim3(NTOK), dim3(256), 0, stream>>>(t1, gl, pj_lg, pj_lb, atmB);
  gemm(atmB, ptmB, BFc, Ec, (float*)d_out, nullptr, pj_b, x2, og8, pj_ogB, pj_ogs, EP_GATE_PJ);
}
